// QincoSubstep_36215164240449
// MI455X (gfx1250) — matrix operand order repaired and checked by reading, not compiled, not run on hardware
//
#include <hip/hip_runtime.h>
#include <stdint.h>

// ---------------- problem constants ----------------
#define B_ROWS   262144
#define DD       128            // feature dim
#define KCODES   256            // codebook entries
#define AA       16             // top-k
#define MT       128            // rows per block
#define RP       136            // padded bf16 row stride (elements) -> 272B rows, kills LDS bank conflicts
#define NBLK     (B_ROWS / MT)  // 2048

// ---------------- CDNA5 WMMA types ----------------
typedef __attribute__((ext_vector_type(16))) __bf16 v16bf;
typedef __attribute__((ext_vector_type(8)))  float  v8f;

union ABFrag { uint4 q[2]; v16bf v; };   // 32B fragment = 8 VGPRs

__device__ __forceinline__ unsigned short f32_to_bf16(float f) {
    union { float f; unsigned u; } c; c.f = f;
    return (unsigned short)((c.u + 0x7FFFu + ((c.u >> 16) & 1u)) >> 16); // RNE
}

// order-preserving map f32 -> u32 (ascending)
__device__ __forceinline__ unsigned fkey(float f) {
    unsigned u = __float_as_uint(f);
    return (u & 0x80000000u) ? ~u : (u | 0x80000000u);
}

// ---------------- LDS layout (dynamic shared) ----------------
// cb_lds  : [KCODES][RP] bf16      69632 B
// res_lds : [MT][RP]     bf16      34816 B
// c2_lds  : [KCODES]     f32        1024 B
// dist_lds: [MT][KCODES] f32      131072 B
// red_lds : [64]         f32         256 B
// code_lds: [MT]         i32         512 B
// xh_lds  : [MT][DD]     f32       65536 B  (kills the 134MB xhat re-read in stage 5)
#define SM_CB    0
#define SM_RES   (SM_CB  + KCODES*RP*2)
#define SM_C2    (SM_RES + MT*RP*2)
#define SM_DIST  (SM_C2  + KCODES*4)
#define SM_RED   (SM_DIST + MT*KCODES*4)
#define SM_CODE  (SM_RED + 64*4)
#define SM_XH    (SM_CODE + MT*4)
#define SM_TOTAL (SM_XH + MT*DD*4)   // = 302848 bytes (< 320KB WGP LDS)

extern "C" __global__ void __launch_bounds__(256)
qinco_main(const float* __restrict__ xhat,
           const float* __restrict__ xtarget,
           const float* __restrict__ last_x,
           const float* __restrict__ cbW,
           int*   __restrict__ idx_out,   // [B][AA]
           float* __restrict__ partial)   // [NBLK]
{
    extern __shared__ char smem[];
    unsigned short* cb_lds   = (unsigned short*)(smem + SM_CB);
    unsigned short* res_lds  = (unsigned short*)(smem + SM_RES);
    float*          c2_lds   = (float*)(smem + SM_C2);
    float*          dist_lds = (float*)(smem + SM_DIST);
    float*          red_lds  = (float*)(smem + SM_RED);
    int*            code_lds = (int*)(smem + SM_CODE);
    float*          xh_lds   = (float*)(smem + SM_XH);

    const int t    = threadIdx.x;
    const int wave = t >> 5;
    const int lane = t & 31;
    const int row0 = blockIdx.x * MT;

    // ---- Stage 1: codebook -> LDS bf16 (+ c2 in f32, matches reference precision) ----
    {
        const float4* crow = (const float4*)(cbW + t * DD);
        float acc = 0.f;
        #pragma unroll 4
        for (int d4 = 0; d4 < DD / 4; ++d4) {
            float4 v = crow[d4];
            acc += v.x*v.x + v.y*v.y + v.z*v.z + v.w*v.w;
            uint2 p;
            p.x = (unsigned)f32_to_bf16(v.x) | ((unsigned)f32_to_bf16(v.y) << 16);
            p.y = (unsigned)f32_to_bf16(v.z) | ((unsigned)f32_to_bf16(v.w) << 16);
            *(uint2*)&cb_lds[t * RP + d4 * 4] = p;
        }
        c2_lds[t] = acc;
    }

    // ---- Stage 2: resid tile -> LDS as bf16(-2*resid); xhat tile cached f32 in LDS ----
    {
        const float4* xh = (const float4*)(xhat    + (size_t)row0 * DD);
        const float4* xt = (const float4*)(xtarget + (size_t)row0 * DD);
        float4*       xc = (float4*)xh_lds;
        #pragma unroll
        for (int i = 0; i < (MT * DD / 4) / 256; ++i) {   // 16 iterations, coalesced b128
            int idx = i * 256 + t;       // float4 index in tile
            int r   = idx >> 5;          // row (DD/4 == 32 float4 per row)
            int d4  = idx & 31;
            float4 a = xt[idx], b = xh[idx];
            xc[idx] = b;                 // keep f32 xhat for the loss pass
            float rx = -2.f * (a.x - b.x), ry = -2.f * (a.y - b.y);
            float rz = -2.f * (a.z - b.z), rw = -2.f * (a.w - b.w);
            uint2 p;
            p.x = (unsigned)f32_to_bf16(rx) | ((unsigned)f32_to_bf16(ry) << 16);
            p.y = (unsigned)f32_to_bf16(rz) | ((unsigned)f32_to_bf16(rw) << 16);
            *(uint2*)&res_lds[r * RP + d4 * 4] = p;
        }
    }
    __syncthreads();

    // ---- Stage 3: score GEMM via v_wmma_f32_16x16x32_bf16 (B double-buffered) ----
    // wave w owns M-subtile rows [w*16, w*16+16) against all 256 codes.
    {
        // A fragment lane layout (ISA 7.12.2, 16-bit A 16x32):
        //   lanes 0-15 : M=lane,    K = {klo..klo+7} in q0, {klo+16..klo+23} in q1, klo=kc*32
        //   lanes 16-31: M=lane-16, same with klo=kc*32+8
        const int mrow = wave * 16 + (lane & 15);
        const int klo  = (lane < 16) ? 0 : 8;
        ABFrag afrag[4];
        #pragma unroll
        for (int kc = 0; kc < 4; ++kc) {
            const unsigned short* ab = &res_lds[mrow * RP + kc * 32 + klo];
            afrag[kc].q[0] = *(const uint4*)(ab);
            afrag[kc].q[1] = *(const uint4*)(ab + 16);
        }
        // B fragment (16-bit B 32x16): same K layout as A: lanes 0-15: N=lane, K={kb..kb+7},{kb+16..kb+23};
        //                              lanes16-31: N=lane-16, kb=kc*32+8
        const int ncol_lane = lane & 15;
        const int kboff     = (lane < 16) ? 0 : 8;
        const int mbase     = wave * 16 + ((lane < 16) ? 0 : 8);

        ABFrag bfr[2][4];
        // prologue: load B fragments for N-tile 0
        {
            const unsigned short* bb = &cb_lds[ncol_lane * RP + kboff];
            #pragma unroll
            for (int kc = 0; kc < 4; ++kc) {
                bfr[0][kc].q[0] = *(const uint4*)(bb + kc * 32);
                bfr[0][kc].q[1] = *(const uint4*)(bb + kc * 32 + 16);
            }
        }
        #pragma unroll
        for (int nt = 0; nt < 16; ++nt) {
            // prefetch B for nt+1 while WMMAs of nt run (hides ds latency)
            if (nt + 1 < 16) {
                const unsigned short* bb =
                    &cb_lds[((nt + 1) * 16 + ncol_lane) * RP + kboff];
                #pragma unroll
                for (int kc = 0; kc < 4; ++kc) {
                    bfr[(nt + 1) & 1][kc].q[0] = *(const uint4*)(bb + kc * 32);
                    bfr[(nt + 1) & 1][kc].q[1] = *(const uint4*)(bb + kc * 32 + 16);
                }
            }
            v8f acc = {};
            #pragma unroll
            for (int kc = 0; kc < 4; ++kc) {
                acc = __builtin_amdgcn_wmma_f32_16x16x32_bf16(
                          false, afrag[kc].v, false, bfr[nt & 1][kc].v,
                          (short)0, acc, false, false);
            }
            // C/D layout: VGPR r -> (M=mbase+r, N=nt*16+ncol_lane)
            const int   ncol = nt * 16 + ncol_lane;
            const float c2v  = c2_lds[ncol];
            #pragma unroll
            for (int r = 0; r < 8; ++r)
                dist_lds[(mbase + r) * KCODES + ncol] = acc[r] + c2v; // x2 dropped: row-constant
        }
    }
    __syncthreads();

    // ---- Stage 4: top-16 per row (wave-parallel repeated argmin, ascending, jax tie-break) ----
    {
        for (int rr = 0; rr < 16; ++rr) {
            const int row = wave * 16 + rr;
            const float4* drow = (const float4*)&dist_lds[row * KCODES + lane * 8];
            float v[8];
            { float4 a = drow[0], b = drow[1];
              v[0]=a.x; v[1]=a.y; v[2]=a.z; v[3]=a.w;
              v[4]=b.x; v[5]=b.y; v[6]=b.z; v[7]=b.w; }
            unsigned used = 0u;
            const int out_base = (row0 + row) * AA;
            for (int it = 0; it < AA; ++it) {
                unsigned long long best = ~0ull;
                #pragma unroll
                for (int j = 0; j < 8; ++j) {
                    if (!((used >> j) & 1u)) {
                        unsigned long long cand =
                            ((unsigned long long)fkey(v[j]) << 32) | (unsigned)(lane * 8 + j);
                        best = (cand < best) ? cand : best;
                    }
                }
                #pragma unroll
                for (int s = 16; s >= 1; s >>= 1) {
                    unsigned long long o = __shfl_xor(best, s, 32);
                    best = (o < best) ? o : best;
                }
                const int n = (int)(best & 0xFFFFFFFFu);
                if ((n >> 3) == lane) used |= 1u << (n & 7);
                if (lane == 0) {
                    idx_out[out_base + it] = n;
                    if (it == 0) code_lds[row] = n;
                }
            }
        }
    }
    __syncthreads();

    // ---- Stage 5: loss partial, full f32 (xhat from LDS, f32 codebook gather) ----
    {
        const float4* xh = (const float4*)xh_lds;
        const float4* lx = (const float4*)(last_x + (size_t)row0 * DD);
        float lsum = 0.f;
        #pragma unroll
        for (int i = 0; i < (MT * DD / 4) / 256; ++i) {
            int idx = i * 256 + t;
            int r   = idx >> 5;
            int d4  = idx & 31;
            int code = code_lds[r];
            float4 a  = xh[idx];
            float4 b  = lx[idx];
            float4 cv = *(const float4*)(cbW + code * DD + d4 * 4);
            float e0 = a.x + cv.x - b.x, e1 = a.y + cv.y - b.y;
            float e2 = a.z + cv.z - b.z, e3 = a.w + cv.w - b.w;
            lsum += e0*e0 + e1*e1 + e2*e2 + e3*e3;
        }
        #pragma unroll
        for (int s = 16; s >= 1; s >>= 1) lsum += __shfl_xor(lsum, s, 32);
        if (lane == 0) red_lds[wave] = lsum;
        __syncthreads();
        if (t == 0) {
            float sum = 0.f;
            #pragma unroll
            for (int w = 0; w < 8; ++w) sum += red_lds[w];
            partial[blockIdx.x] = sum;
        }
    }
}

extern "C" __global__ void __launch_bounds__(256)
qinco_finalize(const float* __restrict__ partial, float* __restrict__ out)
{
    __shared__ float s[256];
    float a = 0.f;
    for (int i = threadIdx.x; i < NBLK; i += 256) a += partial[i];
    s[threadIdx.x] = a;
    __syncthreads();
    for (int w = 128; w > 0; w >>= 1) {
        if (threadIdx.x < (unsigned)w) s[threadIdx.x] += s[threadIdx.x + w];
        __syncthreads();
    }
    if (threadIdx.x == 0)
        out[0] = s[0] * (1.0f / ((float)B_ROWS * (float)DD));
}

extern "C" void kernel_launch(void* const* d_in, const int* in_sizes, int n_in,
                              void* d_out, int out_size, void* d_ws, size_t ws_size,
                              hipStream_t stream) {
    const float* xhat    = (const float*)d_in[0];
    const float* xtarget = (const float*)d_in[1];
    const float* last_x  = (const float*)d_in[2];
    const float* cbW     = (const float*)d_in[3];

    float* out     = (float*)d_out;        // out[0] = loss
    int*   idx_out = (int*)(out + 1);      // then idx_BA flat [B][16] (int32 bits)
    float* partial = (float*)d_ws;         // NBLK f32 partial sums (8 KB)

    // ~296 KB dynamic LDS: within the 320 KB/WGP CDNA5 budget (1 workgroup/WGP).
    qinco_main<<<NBLK, 256, SM_TOTAL, stream>>>(xhat, xtarget, last_x, cbW, idx_out, partial);
    qinco_finalize<<<1, 256, 0, stream>>>(partial, out);
}
